// MatAttention_33904471835206
// MI455X (gfx1250) — matrix operand order repaired and checked by reading, not compiled, not run on hardware
//
#include <hip/hip_runtime.h>

typedef __attribute__((ext_vector_type(16))) _Float16 v16h;
typedef __attribute__((ext_vector_type(8)))  _Float16 v8h;
typedef __attribute__((ext_vector_type(8)))  float    v8f;

#define SS 512
#define DH 64
#define NH 16
#define NB 16

static __device__ __forceinline__ v8f wmma_f16(v16h a, v16h b, v8f c) {
  // D = A(16x32 f16) * B(32x16 f16) + C(16x16 f32)
  return __builtin_amdgcn_wmma_f32_16x16x32_f16(false, a, false, b, (short)0, c,
                                                false, false);
}

// elements 0..7 from 8 consecutive floats at p0, 8..15 from p1 (both 16B aligned)
static __device__ __forceinline__ v16h cvt8x2(const float* p0, const float* p1) {
  float4 a = *(const float4*)(p0);
  float4 b = *(const float4*)(p0 + 4);
  float4 c = *(const float4*)(p1);
  float4 d = *(const float4*)(p1 + 4);
  v16h r;
  r[0]=(_Float16)a.x;  r[1]=(_Float16)a.y;  r[2]=(_Float16)a.z;  r[3]=(_Float16)a.w;
  r[4]=(_Float16)b.x;  r[5]=(_Float16)b.y;  r[6]=(_Float16)b.z;  r[7]=(_Float16)b.w;
  r[8]=(_Float16)c.x;  r[9]=(_Float16)c.y;  r[10]=(_Float16)c.z; r[11]=(_Float16)c.w;
  r[12]=(_Float16)d.x; r[13]=(_Float16)d.y; r[14]=(_Float16)d.z; r[15]=(_Float16)d.w;
  return r;
}

// B fragment of QK^T: 16 consecutive floats of one K-row (K=16*lane_hi+e)
static __device__ __forceinline__ v16h loadB16(const float* p) {
  return cvt8x2(p, p + 8);
}

// ---------------------------------------------------------------------------
// Kernel 1: per (b,q) row stats: 1/(sum(adj)+eps), max(-dist_masked), 1/sum(exp)
// ---------------------------------------------------------------------------
__global__ __launch_bounds__(128)
void row_stats_kernel(const float* __restrict__ adj, const float* __restrict__ dist,
                      const float* __restrict__ mask, float4* __restrict__ stats) {
  const int row = blockIdx.x;          // b*512 + q
  const int b   = row >> 9;
  const float* arow = adj  + (long)row * SS;
  const float* drow = dist + (long)row * SS;
  const float* mrow = mask + (long)b * SS;
  const int tid = threadIdx.x;

  float asum = 0.f;
  float nmax = -3.0e38f;
  float nd[4];
#pragma unroll
  for (int i = 0; i < 4; ++i) {
    int k = tid + (i << 7);
    float a = arow[k], d = drow[k], mm = mrow[k];
    asum += a;
    float v = (mm == 0.f) ? -3.0e38f : -d;
    nd[i] = v;
    nmax = fmaxf(nmax, v);
  }
#pragma unroll
  for (int o = 16; o >= 1; o >>= 1) {
    asum += __shfl_xor(asum, o, 32);
    nmax  = fmaxf(nmax, __shfl_xor(nmax, o, 32));
  }
  __shared__ float red[24];
  const int wid = tid >> 5, lid = tid & 31;
  if (lid == 0) { red[wid] = asum; red[8 + wid] = nmax; }
  __syncthreads();
  float As = red[0] + red[1] + red[2] + red[3];
  float Nm = fmaxf(fmaxf(red[8], red[9]), fmaxf(red[10], red[11]));
  float es = 0.f;
#pragma unroll
  for (int i = 0; i < 4; ++i)
    if (nd[i] > -1.0e37f) es += __expf(nd[i] - Nm);
#pragma unroll
  for (int o = 16; o >= 1; o >>= 1) es += __shfl_xor(es, o, 32);
  if (lid == 0) red[16 + wid] = es;
  __syncthreads();
  if (tid == 0) {
    float Es = red[16] + red[17] + red[18] + red[19];
    stats[row] = make_float4(1.f / (As + 1e-6f), Nm, 1.f / Es, 0.f);
  }
}

// ---------------------------------------------------------------------------
// Kernel 2: fused attention. 8 waves/block, each wave owns 16 query rows.
// Pass 1: QK^T via WMMA + PER-LANE online softmax (VALU only), one 4-stage
//         (m,s) merge butterfly at the end (64 ds_bpermute/wave total).
// Pass 2: recompute scores, emit p_attn, blend with p_dist/adj_norm, stage f16
//         PW tiles through LDS into A-fragment layout; V staged cooperatively
//         per k-chunk as transposed f16 in LDS (2x ds_load_b128 per fragment).
// ---------------------------------------------------------------------------
__global__ __launch_bounds__(256)
void attn_kernel(const float* __restrict__ Q, const float* __restrict__ K,
                 const float* __restrict__ V, const float* __restrict__ mask,
                 const float* __restrict__ adj, const float* __restrict__ dist,
                 const float4* __restrict__ stats,
                 float* __restrict__ out, float* __restrict__ pattn) {
  __shared__ _Float16 lds_pw[8][16][32];   // per-wave 16x32 f16 PW staging (8 KB)
  __shared__ _Float16 vt[64][40];          // V^T f16 tile [n][kk], padded (5 KB)

  const int bid = blockIdx.x;     // h fastest -> adj/dist stay hot in L2
  const int h   = bid & 15;
  const int qc  = (bid >> 4) & 3;
  const int b   = bid >> 6;
  const int tid = threadIdx.x;
  const int w   = tid >> 5;
  const int lane = tid & 31;
  const int ll  = lane & 15;      // column / row-in-tile lane index
  const int lh  = lane >> 4;      // which 16-lane half
  const int q0  = qc * 128 + w * 16;

  const long bh = (long)(b * NH + h);
  const float*  Qb    = Q    + bh * SS * DH;
  const float*  Kb    = K    + bh * SS * DH;
  const float*  Vb    = V    + bh * SS * DH;
  const float*  maskb = mask + (long)b * SS;
  const float*  adjb  = adj  + (long)b * SS * SS;
  const float*  distb = dist + (long)b * SS * SS;
  const float4* statsb= stats+ (long)b * SS;
  float* pb = pattn + bh * SS * SS;
  float* ob = out   + bh * SS * DH;

  // A fragments of Q (row m = ll), two K=32 chunks covering D=64
  const float* qrow = Qb + (long)(q0 + ll) * DH;
  v16h qf[2];
#pragma unroll
  for (int c = 0; c < 2; ++c)
    qf[c] = cvt8x2(qrow + 32 * c + 16 * lh, qrow + 32 * c + 16 * lh + 8);

  // ---- Pass 1: per-lane online softmax over the lane's own columns ----
  float pm[8], ps[8];
#pragma unroll
  for (int v = 0; v < 8; ++v) { pm[v] = -3.0e38f; ps[v] = 0.f; }

  for (int t = 0; t < 32; ++t) {
    const float* krow = Kb + (long)(16 * t + ll) * DH;
    v8f acc = {};
    acc = wmma_f16(qf[0], loadB16(krow + 16 * lh), acc);
    acc = wmma_f16(qf[1], loadB16(krow + 32 + 16 * lh), acc);
    float mk = maskb[16 * t + ll];
#pragma unroll
    for (int v = 0; v < 8; ++v) {
      float sv = (mk == 0.f) ? -1.0e12f : acc[v] * 0.125f;
      float nm = fmaxf(pm[v], sv);
      ps[v] = ps[v] * __expf(pm[v] - nm) + __expf(sv - nm);
      pm[v] = nm;
    }
  }
  // merge (m,s) pairs across the 16 lanes of each half (4 butterfly stages)
  float m8[8], is8[8];
#pragma unroll
  for (int v = 0; v < 8; ++v) {
    float m = pm[v], s = ps[v];
#pragma unroll
    for (int o = 8; o >= 1; o >>= 1) {
      float mo = __shfl_xor(m, o, 32);
      float so = __shfl_xor(s, o, 32);
      float M  = fmaxf(m, mo);
      s = s * __expf(m - M) + so * __expf(mo - M);
      m = M;
    }
    m8[v] = m;
    is8[v] = 1.f / s;
  }

  float iadj[8], md[8], isd[8];
#pragma unroll
  for (int v = 0; v < 8; ++v) {
    float4 st = statsb[q0 + v + 8 * lh];
    iadj[v] = st.x; md[v] = st.y; isd[v] = st.z;
  }

  // ---- Pass 2: recompute, emit p_attn, blend, P*V ----
  v8f oacc[4] = {};
  for (int kc = 0; kc < 16; ++kc) {
    // cooperative stage of V^T tile (32 rows x 64 cols -> vt[n][kk], f16)
    __syncthreads();                       // WAR: previous chunk's reads done
    {
      const float* vsrc = Vb + (long)(32 * kc) * DH;
#pragma unroll
      for (int j = 0; j < 8; ++j) {
        int e = tid + (j << 8);            // coalesced
        vt[e & 63][e >> 6] = (_Float16)vsrc[e];
      }
    }
    __syncthreads();                       // RAW: tile visible to all waves

#pragma unroll
    for (int tt = 0; tt < 2; ++tt) {
      const int t = 2 * kc + tt;
      const float* krow = Kb + (long)(16 * t + ll) * DH;
      v8f acc = {};
      acc = wmma_f16(qf[0], loadB16(krow + 16 * lh), acc);
      acc = wmma_f16(qf[1], loadB16(krow + 32 + 16 * lh), acc);
      const int col = 16 * t + ll;
      const float mk = maskb[col];
#pragma unroll
      for (int v = 0; v < 8; ++v) {
        int qi = q0 + v + 8 * lh;
        float sv = (mk == 0.f) ? -1.0e12f : acc[v] * 0.125f;
        float p  = __expf(sv - m8[v]) * is8[v];
        pb[(long)qi * SS + col] = p;
        float av = adjb[(long)qi * SS + col];
        float dv = distb[(long)qi * SS + col];
        float pd = (mk == 0.f) ? 0.f : __expf(-dv - md[v]) * isd[v];
        float pw = 0.33f * p + 0.33f * pd + 0.34f * (av * iadj[v]);
        lds_pw[w][v + 8 * lh][(tt << 4) + ll] = (_Float16)pw;
      }
    }

    asm volatile("s_wait_dscnt 0x0" ::: "memory");  // LDS store->load x-lane hazard
    v8h plo = *(const v8h*)&lds_pw[w][ll][16 * lh];
    v8h phi = *(const v8h*)&lds_pw[w][ll][16 * lh + 8];
    v16h af;
#pragma unroll
    for (int i = 0; i < 8; ++i) { af[i] = plo[i]; af[8 + i] = phi[i]; }

#pragma unroll
    for (int nt = 0; nt < 4; ++nt) {
      v8h vlo = *(const v8h*)&vt[16 * nt + ll][16 * lh];
      v8h vhi = *(const v8h*)&vt[16 * nt + ll][16 * lh + 8];
      v16h vf;
#pragma unroll
      for (int i = 0; i < 8; ++i) { vf[i] = vlo[i]; vf[8 + i] = vhi[i]; }
      oacc[nt] = wmma_f16(af, vf, oacc[nt]);
    }
  }

#pragma unroll
  for (int nt = 0; nt < 4; ++nt)
#pragma unroll
    for (int v = 0; v < 8; ++v)
      ob[(long)(q0 + v + 8 * lh) * DH + 16 * nt + ll] = oacc[nt][v];
}

// ---------------------------------------------------------------------------
extern "C" void kernel_launch(void* const* d_in, const int* in_sizes, int n_in,
                              void* d_out, int out_size, void* d_ws, size_t ws_size,
                              hipStream_t stream) {
  const float* Q    = (const float*)d_in[0];
  const float* K    = (const float*)d_in[1];
  const float* V    = (const float*)d_in[2];
  const float* mask = (const float*)d_in[3];
  const float* adj  = (const float*)d_in[4];
  const float* dist = (const float*)d_in[5];

  float* out   = (float*)d_out;
  float* pattn = out + (long)NB * NH * SS * DH;   // second tuple element
  float4* stats = (float4*)d_ws;                  // 16*512*16B = 128 KB

  row_stats_kernel<<<NB * SS, 128, 0, stream>>>(adj, dist, mask, stats);
  attn_kernel<<<NB * NH * (SS / 128), 256, 0, stream>>>(Q, K, V, mask, adj, dist,
                                                        stats, out, pattn);
}
